// DenseKAN_31568009625702
// MI455X (gfx1250) — compile-verified, matrix operand order checked by reading, not run on hardware
//
#include <hip/hip_runtime.h>
#include <hip/hip_bf16.h>

// ---------------------------------------------------------------------------
// DenseKAN forward as a single fused-weight GEMM on CDNA5 WMMA.
//   out[b,o] = sum_{i,k} Aaug[b, i*9+k] * Wf[i*9+k, o] + bias[o]
//   Aaug[b,i*9+k] = bases_k(x[b,i])        (k<8)     |  silu(x[b,i])   (k==8)
//   Wf[i*9+k, o]  = spline_kernel[i,k,o]*scale[i,o]  |  scale[i,o]     (k==8)
// M=1024, N=512, K=4608. f16 operands, f32 accumulation.
// Tile staging: A via TDM tensor_load_to_lds (TENSORcnt), B via
// global_load_async_to_lds_b128 (ASYNCcnt).
// ---------------------------------------------------------------------------

#define BATCH    1024
#define IN_SIZE  512
#define UNITS    512
#define KDIM     (IN_SIZE * 9)      // 4608 halves = 9216 B = 2304 dwords per row

#define BM 128
#define BN 64
#define BK 32
#define LDSH 40                     // halves per LDS row (80B = 64B data + 16B pad)

typedef __attribute__((ext_vector_type(16))) _Float16     v16h;
typedef __attribute__((ext_vector_type(8)))  _Float16     v8h;
typedef __attribute__((ext_vector_type(8)))  float        v8f;
typedef __attribute__((ext_vector_type(4)))  unsigned int v4u;
typedef __attribute__((ext_vector_type(8)))  int          v8i;
typedef __attribute__((ext_vector_type(4)))  int          v4i;

#if defined(__has_builtin)
#  if __has_builtin(__builtin_amdgcn_tensor_load_to_lds)
#    define HAVE_TDM 1
#  endif
#endif
#ifndef HAVE_TDM
#  define HAVE_TDM 0
#endif

// ---------------------------------------------------------------------------
// Phase 1a: cubic B-spline bases (Cox-de Boor, uniform knots t_j = -2.2+0.4j)
// plus silu(x) appended as the 9th basis, written as f16 row-major A (1024x4608).
// ---------------------------------------------------------------------------
__global__ __launch_bounds__(256) void kan_bases_kernel(const float* __restrict__ x,
                                                        _Float16* __restrict__ A)
{
    const int tid = blockIdx.x * blockDim.x + threadIdx.x;   // tid = b*IN_SIZE + i
    if (tid >= BATCH * IN_SIZE) return;
    const float xv = x[tid];
    const float h = 0.4f;

    float b[11];
#pragma unroll
    for (int j = 0; j < 11; ++j) {
        float t0 = -2.2f + h * (float)j;
        b[j] = (xv >= t0 && xv < t0 + h) ? 1.0f : 0.0f;      // order-0 indicators
    }
#pragma unroll
    for (int k = 1; k <= 3; ++k) {
        const float inv = 1.0f / (h * (float)k);
#pragma unroll
        for (int j = 0; j + k < 11; ++j) {                   // in-place ascending recursion
            float tj = -2.2f + h * (float)j;
            float tr = tj + h * (float)(k + 1);              // t_{j+k+1}
            b[j] = (xv - tj) * inv * b[j] + (tr - xv) * inv * b[j + 1];
        }
    }
    const float s = xv / (1.0f + __expf(-xv));               // silu

    const int bi = tid >> 9;
    const int i  = tid & (IN_SIZE - 1);
    _Float16* row = A + (size_t)bi * KDIM + i * 9;
#pragma unroll
    for (int k = 0; k < 8; ++k) row[k] = (_Float16)b[k];
    row[8] = (_Float16)s;
}

// ---------------------------------------------------------------------------
// Phase 1b: fused weights, stored transposed (N-major: Bt[o][ik]) so that
// B-matrix WMMA fragments load with the same K-contiguous pattern as A.
// ---------------------------------------------------------------------------
__global__ __launch_bounds__(256) void kan_weights_kernel(const float* __restrict__ sk,
                                                          const float* __restrict__ sf,
                                                          _Float16* __restrict__ Bt)
{
    const int tid = blockIdx.x * blockDim.x + threadIdx.x;   // tid = i*UNITS + o
    if (tid >= IN_SIZE * UNITS) return;
    const int i = tid >> 9;
    const int o = tid & (UNITS - 1);
    const float scale = sf[tid];
    _Float16* row = Bt + (size_t)o * KDIM + i * 9;
#pragma unroll
    for (int k = 0; k < 8; ++k)
        row[k] = (_Float16)(sk[(size_t)((i << 3) + k) * UNITS + o] * scale);
    row[8] = (_Float16)scale;
}

// ---------------------------------------------------------------------------
// Phase 2: WMMA GEMM. Block tile 128x64, 8 waves, each wave computes 32x32
// (2x2 v_wmma_f32_16x16x32_f16). K stepped by 32 through LDS; A tile DMA'd by
// the Tensor Data Mover (with native LDS padding to an 80B row stride), B tile
// streamed by per-lane async global->LDS loads.
// ---------------------------------------------------------------------------
__device__ inline v16h frag_combine(v8h lo, v8h hi)
{
    v16h r;
#pragma unroll
    for (int e = 0; e < 8; ++e) { r[e] = lo[e]; r[e + 8] = hi[e]; }
    return r;
}

__global__ __launch_bounds__(256) void kan_gemm_kernel(const _Float16* __restrict__ A,
                                                       const _Float16* __restrict__ Bt,
                                                       const float* __restrict__ bias,
                                                       float* __restrict__ out)
{
    __shared__ _Float16 sA[BM * LDSH];   // 128 rows x 80B (10240 B)
    __shared__ _Float16 sB[BN * LDSH];   //  64 rows x 80B ( 5120 B)

    const int tid  = threadIdx.x;
    const int lane = tid & 31;
    const int wave = tid >> 5;           // 0..7
    const int wm   = wave >> 1;          // 0..3 (M direction)
    const int wn   = wave & 1;           // 0..1 (N direction)
    const int rowBase = blockIdx.y * BM;
    const int colBase = blockIdx.x * BN;

    // LDS byte offsets (flat LDS address truncates to the LDS offset, ISA 10.2).
    const unsigned sA_lds = (unsigned)(unsigned long long)(void*)&sA[0];
    const unsigned sB_lds = (unsigned)(unsigned long long)(void*)&sB[0];

    // Per-lane fragment addressing (ISA 16-bit A/B layout):
    // lanes 0-15 hold K chunks {0..7}U{16..23}; lanes 16-31 hold {8..15}U{24..31}.
    const int fr = lane & 15;
    const int kb = (lane >> 4) << 3;     // 0 or 8

    v8f acc[2][2];
#pragma unroll
    for (int mi = 0; mi < 2; ++mi)
#pragma unroll
        for (int ni = 0; ni < 2; ++ni)
#pragma unroll
            for (int e = 0; e < 8; ++e) acc[mi][ni][e] = 0.0f;

    // B tile: 64 rows x 4 chunks(16B) = 256 chunks -> one per thread.
    const int br = tid >> 2, bc = tid & 3;
    const unsigned       ldsB   = sB_lds + (unsigned)(br * LDSH + bc * 8) * 2u;
    const _Float16*      gBbase = Bt + (size_t)(colBase + br) * KDIM + bc * 8;

#if !HAVE_TDM
    // Fallback A staging: 512 chunks -> two per thread, via async loads.
    const int ar0 = tid >> 2, ac = tid & 3, ar1 = (tid >> 2) + 64;
    const unsigned  ldsA0 = sA_lds + (unsigned)(ar0 * LDSH + ac * 8) * 2u;
    const unsigned  ldsA1 = sA_lds + (unsigned)(ar1 * LDSH + ac * 8) * 2u;
    const _Float16* gA0   = A + (size_t)(rowBase + ar0) * KDIM + ac * 8;
    const _Float16* gA1   = A + (size_t)(rowBase + ar1) * KDIM + ac * 8;
#endif

#pragma unroll 1
    for (int k0 = 0; k0 < KDIM; k0 += BK) {
        if (k0 + BK < KDIM)
            __builtin_prefetch(gBbase + k0 + BK, 0, 3);      // global_prefetch_b8

        // ---- B tile: per-lane async global->LDS (ASYNCcnt) ----
        {
            unsigned long long ga = (unsigned long long)(const void*)(gBbase + k0);
            asm volatile("global_load_async_to_lds_b128 %0, %1, off"
                         :: "v"(ldsB), "v"(ga) : "memory");
        }

        // ---- A tile: Tensor Data Mover (TENSORcnt) ----
#if HAVE_TDM
        if (wave == 0) {
            unsigned long long ga =
                (unsigned long long)(const void*)(A + (size_t)rowBase * KDIM + k0);
            // D# group 0: count=1 | lds_addr | global_addr(57b) | type=2
            v4u g0 = { 1u,
                       sA_lds,
                       (unsigned)ga,
                       (unsigned)((ga >> 32) & 0x01FFFFFFu) | (2u << 30) };
            // D# group 1: data_size=4B, pad 4 dwords every 16 dwords (64B -> 80B rows),
            // tensor 2304 x 1024 dwords, tile 16 x 128 dwords, row stride 2304 dwords.
            v8i g1 = { (int)((2u << 16) | (1u << 20) | (3u << 22) | (3u << 25)),
                       (int)(((KDIM / 2) & 0xFFFFu) << 16),          // tensor_dim0 lo16
                       (int)(((KDIM / 2) >> 16) | ((unsigned)BATCH << 16)), // dim0 hi | dim1 lo
                       (int)(16u << 16),                             // dim1 hi(0) | tile_dim0
                       (int)(BM),                                    // tile_dim1 | tile_dim2=0
                       (int)(KDIM / 2),                              // tensor_dim0_stride lo32
                       0, 0 };
            v4i gz = { 0, 0, 0, 0 };
#if __clang_major__ >= 23
            v8i gp = { 0, 0, 0, 0, 0, 0, 0, 0 };
            __builtin_amdgcn_tensor_load_to_lds(g0, g1, gz, gz, gp, 0);
#else
            __builtin_amdgcn_tensor_load_to_lds(g0, g1, gz, gz, 0);
#endif
        }
#else
        {
            unsigned long long ga0 = (unsigned long long)(const void*)(gA0 + k0);
            unsigned long long ga1 = (unsigned long long)(const void*)(gA1 + k0);
            asm volatile("global_load_async_to_lds_b128 %0, %1, off"
                         :: "v"(ldsA0), "v"(ga0) : "memory");
            asm volatile("global_load_async_to_lds_b128 %0, %1, off"
                         :: "v"(ldsA1), "v"(ga1) : "memory");
        }
#endif

        // ---- complete DMA, publish LDS to all waves ----
        asm volatile("s_wait_asynccnt 0" ::: "memory");
#if HAVE_TDM
        if (wave == 0) __builtin_amdgcn_s_wait_tensorcnt(0);
#endif
        __syncthreads();

        // ---- fragments + WMMA ----
        v16h afrag[2], bfrag[2];
#pragma unroll
        for (int s = 0; s < 2; ++s) {
            const int ra = wm * 32 + s * 16 + fr;
            afrag[s] = frag_combine(*(const v8h*)(&sA[ra * LDSH + kb]),
                                    *(const v8h*)(&sA[ra * LDSH + kb + 16]));
            const int rb = wn * 32 + s * 16 + fr;
            bfrag[s] = frag_combine(*(const v8h*)(&sB[rb * LDSH + kb]),
                                    *(const v8h*)(&sB[rb * LDSH + kb + 16]));
        }
#pragma unroll
        for (int mi = 0; mi < 2; ++mi)
#pragma unroll
            for (int ni = 0; ni < 2; ++ni)
                acc[mi][ni] = __builtin_amdgcn_wmma_f32_16x16x32_f16(
                    false, afrag[mi], false, bfrag[ni],
                    (short)0, acc[mi][ni], false, false);
        __syncthreads();
    }

    // Epilogue: C/D layout -> lane (L&15) = N, VGPR r holds M = r + 8*(L>=16).
#pragma unroll
    for (int mi = 0; mi < 2; ++mi) {
#pragma unroll
        for (int ni = 0; ni < 2; ++ni) {
            const int col   = colBase + wn * 32 + ni * 16 + fr;
            const int mrow0 = rowBase + wm * 32 + mi * 16 + ((lane >> 4) << 3);
            const float bv  = bias[col];
#pragma unroll
            for (int r = 0; r < 8; ++r)
                out[(size_t)(mrow0 + r) * UNITS + col] = acc[mi][ni][r] + bv;
        }
    }
}

// ---------------------------------------------------------------------------
// Launch: three kernels on `stream`; scratch holds f16 A then f16 Bt.
// ---------------------------------------------------------------------------
extern "C" void kernel_launch(void* const* d_in, const int* in_sizes, int n_in,
                              void* d_out, int out_size, void* d_ws, size_t ws_size,
                              hipStream_t stream)
{
    const float* x    = (const float*)d_in[0];   // (1024, 512)
    const float* sk   = (const float*)d_in[1];   // (512, 8, 512)
    const float* sf   = (const float*)d_in[2];   // (512, 512)
    const float* bias = (const float*)d_in[3];   // (512,)
    float* out = (float*)d_out;                  // (1024, 512)

    _Float16* Amat = (_Float16*)d_ws;                                     // 9.0 MB
    _Float16* Bmat = (_Float16*)((char*)d_ws + (size_t)BATCH * KDIM * 2); // 4.5 MB

    kan_bases_kernel<<<(BATCH * IN_SIZE) / 256, 256, 0, stream>>>(x, Amat);
    kan_weights_kernel<<<(IN_SIZE * UNITS) / 256, 256, 0, stream>>>(sk, sf, Bmat);
    kan_gemm_kernel<<<dim3(UNITS / BN, BATCH / BM), 256, 0, stream>>>(Amat, Bmat, bias, out);
}
